// TFXLMMainLayer_35613868819150
// MI455X (gfx1250) — matrix operand order repaired and checked by reading, not compiled, not run on hardware
//
#include <hip/hip_runtime.h>
#include <hip/hip_bf16.h>

typedef __bf16 bf16_t;
typedef __attribute__((ext_vector_type(16))) __bf16 v16bf;
typedef __attribute__((ext_vector_type(8)))  float  v8f;

#define NB    8
#define NS    512
#define ND    1024
#define NH    16
#define NDH   64
#define NL    12
#define NFF   4096
#define NTOK  (NB * NS)   // 4096 rows

#define WMMA_BF16(a, b, c) \
  __builtin_amdgcn_wmma_f32_16x16x32_bf16(false, (a), false, (b), (short)0, (c), false, false)

// ---------------- WMMA fragment loaders (layouts per ISA 7.12.2) ----------------
// All loads are two 16-byte vectors; callers guarantee 16B alignment.

// A fragment 16x32 bf16 from row-major [rows][K]: element(row,k) = m[(row0+row)*rstride + k]
// lane layout: row = lane&15; VGPR p<4 -> K=2p,2p+1 ; p>=4 -> K=16+2(p-4)... ; lanes>=16: K+8
static __device__ __forceinline__ v16bf ld_afrag(const bf16_t* m, int row0, int rstride) {
  const int lane = threadIdx.x & 31;
  const int r    = row0 + (lane & 15);
  const int koff = (lane >> 4) << 3;
  const bf16_t* base = m + (size_t)r * rstride + koff;
  union { uint4 u[2]; v16bf v; } t;
  t.u[0] = *reinterpret_cast<const uint4*>(base);        // K = koff .. koff+7
  t.u[1] = *reinterpret_cast<const uint4*>(base + 16);   // K = koff+16 .. koff+23
  return t.v;
}

// B fragment 32x16 bf16 from a K-contiguous (transposed) source:
// element(k, c) = m[(col0+c)*rstride + kbase + k]
// lane layout: col = lane&15; lanes 0-15 K=0..7,16..23, lanes 16-31 K=8..15,24..31 (2 per VGPR)
static __device__ __forceinline__ v16bf ld_bfragT(const bf16_t* m, int col0, int rstride, int kbase) {
  const int lane = threadIdx.x & 31;
  const int c    = col0 + (lane & 15);
  const int koff = (lane >> 4) << 3;
  const bf16_t* base = m + (size_t)c * rstride + kbase + koff;
  union { uint4 u[2]; v16bf v; } t;
  t.u[0] = *reinterpret_cast<const uint4*>(base);        // K = koff .. koff+7
  t.u[1] = *reinterpret_cast<const uint4*>(base + 16);   // K = koff+16 .. koff+23
  return t.v;
}

// ---------------- fp32 -> bf16 transpose-convert (weights -> [N][K]) ----------------

__global__ __launch_bounds__(256) void conv_transpose_kernel(
    const float* __restrict__ in, bf16_t* __restrict__ out, int K, int Nn) {
  __shared__ bf16_t t[32][33];
  const int tx = threadIdx.x & 31;
  const int ty = threadIdx.x >> 5;  // 0..7
  const int k0 = blockIdx.y * 32;
  const int n0 = blockIdx.x * 32;
#pragma unroll
  for (int j = 0; j < 4; ++j)
    t[ty + 8 * j][tx] = (bf16_t)in[(size_t)(k0 + ty + 8 * j) * Nn + n0 + tx];
  __syncthreads();
#pragma unroll
  for (int j = 0; j < 4; ++j)
    out[(size_t)(n0 + ty + 8 * j) * K + k0 + tx] = t[tx][ty + 8 * j];
}

// ---------------- bf16 GEMM: C[M,Nn] = A[M,K] * Bt[Nn,K]^T + bias (+epilogue) -------
// Block: 256 threads = 8 waves (4x2), tile 128x128, K-step 32. Each wave: 32x64 (2x4 tiles).

#define BM 128
#define BN 128
#define BK 32
#define LDT (BK + 8)   // 40 elems = 80B rows -> 16B aligned fragment bases

__global__ __launch_bounds__(256) void gemm_bf16_kernel(
    const bf16_t* __restrict__ A, const bf16_t* __restrict__ Bt,
    const float* __restrict__ bias, const float* __restrict__ resid,
    float* __restrict__ outF, bf16_t* __restrict__ outB, bf16_t* __restrict__ outVT,
    int M, int K, int Nn, float scale, int doGelu) {
  __shared__ bf16_t As[BM][LDT];
  __shared__ bf16_t Bs[BN][LDT];

  const int tid   = threadIdx.x;
  const int wave  = tid >> 5;
  const int lane  = tid & 31;
  const int waveM = (wave >> 1) * 32;
  const int waveN = (wave & 1) * 64;
  const int m0    = blockIdx.y * BM;
  const int n0    = blockIdx.x * BN;

  v8f acc[2][4];
#pragma unroll
  for (int mt = 0; mt < 2; ++mt)
#pragma unroll
    for (int nt = 0; nt < 4; ++nt) acc[mt][nt] = {};

  const int srow = tid >> 1, scg = (tid & 1) * 16;  // 128 rows x 32 K, 16 elems/thread

  const bf16_t* agp = A  + (size_t)(m0 + srow) * K + scg;
  const bf16_t* bgp = Bt + (size_t)(n0 + srow) * K + scg;

  for (int k0 = 0; k0 < K; k0 += BK) {
    const uint4* ag = reinterpret_cast<const uint4*>(agp + k0);
    uint4 av0 = ag[0], av1 = ag[1];
    const uint4* bg = reinterpret_cast<const uint4*>(bgp + k0);
    uint4 bv0 = bg[0], bv1 = bg[1];

    if (k0 + BK < K) {
      __builtin_prefetch(agp + k0 + BK, 0, 1);
      __builtin_prefetch(bgp + k0 + BK, 0, 1);
    }

    __syncthreads();
    *reinterpret_cast<uint4*>(&As[srow][scg])     = av0;
    *reinterpret_cast<uint4*>(&As[srow][scg + 8]) = av1;
    *reinterpret_cast<uint4*>(&Bs[srow][scg])     = bv0;
    *reinterpret_cast<uint4*>(&Bs[srow][scg + 8]) = bv1;
    __syncthreads();

    v16bf af[2], bf_[4];
#pragma unroll
    for (int mt = 0; mt < 2; ++mt) af[mt]  = ld_afrag(&As[0][0], waveM + mt * 16, LDT);
#pragma unroll
    for (int nt = 0; nt < 4; ++nt) bf_[nt] = ld_bfragT(&Bs[0][0], waveN + nt * 16, LDT, 0);
#pragma unroll
    for (int mt = 0; mt < 2; ++mt)
#pragma unroll
      for (int nt = 0; nt < 4; ++nt) acc[mt][nt] = WMMA_BF16(af[mt], bf_[nt], acc[mt][nt]);
  }

  const int halfr = (lane >> 4) << 3;
  const int cl    = lane & 15;
#pragma unroll
  for (int mt = 0; mt < 2; ++mt)
#pragma unroll
    for (int nt = 0; nt < 4; ++nt) {
      const int gcol = n0 + waveN + nt * 16 + cl;
      const float bvv = bias[gcol];
#pragma unroll
      for (int i = 0; i < 8; ++i) {
        const int grow = m0 + waveM + mt * 16 + i + halfr;
        float v = acc[mt][nt][i] + bvv;
        if (doGelu) v = 0.5f * v * (1.0f + erff(v * 0.70710678f));
        v *= scale;
        const size_t off = (size_t)grow * Nn + gcol;
        if (resid) v += resid[off];
        if (outF) outF[off] = v;
        if (outB) outB[off] = (bf16_t)v;
        if (outVT) {  // per-head transposed V: vt[b][h][dh][S]
          const int bb = grow >> 9;        // /NS
          const int ss = grow & (NS - 1);
          const int hh = gcol >> 6;        // /NDH
          const int dd = gcol & (NDH - 1);
          outVT[(((size_t)bb * NH + hh) * NDH + dd) * NS + ss] = (bf16_t)v;
        }
      }
    }
}

// ---------------- flash attention: 1 wave = 16 query rows of one (b,h) ----------------
// Block: 128 threads (4 waves) -> 64 query rows. Grid: B*H*(S/64).

__global__ __launch_bounds__(128) void attn_kernel(
    const bf16_t* __restrict__ q, const bf16_t* __restrict__ k,
    const bf16_t* __restrict__ vt, const float* __restrict__ am,
    bf16_t* __restrict__ ctx) {
  __shared__ bf16_t Pb[4][16][40];  // per-wave P staging (16 rows x 32 keys), 80B rows

  const int tid  = threadIdx.x;
  const int w    = tid >> 5;
  const int lane = tid & 31;
  const int blk  = blockIdx.x;
  const int qg64 = blk % (NS / 64);
  const int h    = (blk / (NS / 64)) % NH;
  const int b    = blk / ((NS / 64) * NH);
  const int q0   = qg64 * 64 + w * 16;

  const bf16_t* qg    = q  + ((size_t)(b * NS + q0)) * ND + h * NDH;
  const bf16_t* kg    = k  + ((size_t)(b * NS)) * ND + h * NDH;
  const bf16_t* vt_bh = vt + ((size_t)(b * NH + h)) * NDH * NS;
  const float*  amb   = am + (size_t)b * NS;

  const v16bf aq0 = ld_afrag(qg, 0, ND);       // q cols d=0..31
  const v16bf aq1 = ld_afrag(qg + 32, 0, ND);  // q cols d=32..63

  v8f c0 = {}, c1 = {}, c2 = {}, c3 = {};
  float mst[8], lst[8];
#pragma unroll
  for (int i = 0; i < 8; ++i) { mst[i] = -3.0e38f; lst[i] = 0.0f; }

  const int cl    = lane & 15;
  const int halfr = (lane >> 4) << 3;

  for (int kc = 0; kc < NS; kc += 32) {
    // scores for 32 keys: two 16x16 tiles, K=64 via two chained WMMAs each
    v8f s0 = {}, s1 = {};
    {
      v16bf blo = ld_bfragT(kg, kc, ND, 0);
      v16bf bhi = ld_bfragT(kg, kc, ND, 32);
      s0 = WMMA_BF16(aq0, blo, s0);
      s0 = WMMA_BF16(aq1, bhi, s0);
      blo = ld_bfragT(kg, kc + 16, ND, 0);
      bhi = ld_bfragT(kg, kc + 16, ND, 32);
      s1 = WMMA_BF16(aq0, blo, s1);
      s1 = WMMA_BF16(aq1, bhi, s1);
    }
    const float neg0 = -1.0e30f * (1.0f - amb[kc + cl]);
    const float neg1 = -1.0e30f * (1.0f - amb[kc + 16 + cl]);

#pragma unroll
    for (int i = 0; i < 8; ++i) {
      float x0 = s0[i] + neg0;
      float x1 = s1[i] + neg1;
      float mx = fmaxf(x0, x1);
#pragma unroll
      for (int d = 8; d >= 1; d >>= 1) mx = fmaxf(mx, __shfl_xor(mx, d, 32));
      const float mnew  = fmaxf(mst[i], mx);
      const float alpha = __expf(mst[i] - mnew);
      const float p0 = __expf(x0 - mnew);
      const float p1 = __expf(x1 - mnew);
      float rs = p0 + p1;
#pragma unroll
      for (int d = 8; d >= 1; d >>= 1) rs += __shfl_xor(rs, d, 32);
      lst[i] = lst[i] * alpha + rs;
      mst[i] = mnew;
      c0[i] *= alpha; c1[i] *= alpha; c2[i] *= alpha; c3[i] *= alpha;
      const int prow = i + halfr;
      Pb[w][prow][cl]      = (bf16_t)p0;
      Pb[w][prow][16 + cl] = (bf16_t)p1;
    }
    asm volatile("s_wait_dscnt 0" ::: "memory");  // P restage visible before A-frag read

    const v16bf pa = ld_afrag(&Pb[w][0][0], 0, 40);
    c0 = WMMA_BF16(pa, ld_bfragT(vt_bh, 0,  NS, kc), c0);
    c1 = WMMA_BF16(pa, ld_bfragT(vt_bh, 16, NS, kc), c1);
    c2 = WMMA_BF16(pa, ld_bfragT(vt_bh, 32, NS, kc), c2);
    c3 = WMMA_BF16(pa, ld_bfragT(vt_bh, 48, NS, kc), c3);
  }

  bf16_t* cg = ctx + ((size_t)(b * NS + q0)) * ND + h * NDH;
#pragma unroll
  for (int i = 0; i < 8; ++i) {
    const float inv = 1.0f / lst[i];
    const int r = i + halfr;
    cg[(size_t)r * ND + cl]      = (bf16_t)(c0[i] * inv);
    cg[(size_t)r * ND + 16 + cl] = (bf16_t)(c1[i] * inv);
    cg[(size_t)r * ND + 32 + cl] = (bf16_t)(c2[i] * inv);
    cg[(size_t)r * ND + 48 + cl] = (bf16_t)(c3[i] * inv);
  }
}

// ---------------- embeddings + LayerNorm + mask ----------------

__global__ __launch_bounds__(256) void embed_kernel(
    const int* __restrict__ ids, const int* __restrict__ lngs, const float* __restrict__ am,
    const float* __restrict__ we, const float* __restrict__ pe, const float* __restrict__ le,
    const float* __restrict__ g, const float* __restrict__ be,
    float* __restrict__ outF, bf16_t* __restrict__ outB) {
  __shared__ float red[256];
  const int t = threadIdx.x;
  const size_t row = blockIdx.x;
  const int spos = (int)(row % NS);
  const int id = ids[row];
  const int lg = lngs[row];

  float vals[4];
  float s = 0.0f, sq = 0.0f;
#pragma unroll
  for (int j = 0; j < 4; ++j) {
    const int d = t + j * 256;
    const float e = we[(size_t)id * ND + d] + pe[(size_t)spos * ND + d] + le[(size_t)lg * ND + d];
    vals[j] = e; s += e; sq += e * e;
  }
  red[t] = s; __syncthreads();
  for (int o = 128; o > 0; o >>= 1) { if (t < o) red[t] += red[t + o]; __syncthreads(); }
  const float mean = red[0] * (1.0f / ND);
  __syncthreads();
  red[t] = sq; __syncthreads();
  for (int o = 128; o > 0; o >>= 1) { if (t < o) red[t] += red[t + o]; __syncthreads(); }
  const float var  = red[0] * (1.0f / ND) - mean * mean;
  const float rstd = rsqrtf(var + 1e-12f);
  const float fm   = am[row];
#pragma unroll
  for (int j = 0; j < 4; ++j) {
    const int d = t + j * 256;
    float o = ((vals[j] - mean) * rstd * g[d] + be[d]) * fm;
    outF[row * ND + d] = o;
    outB[row * ND + d] = (bf16_t)o;
  }
}

// ---------------- LayerNorm (input already has residual added) ----------------

__global__ __launch_bounds__(256) void ln_kernel(
    const float* __restrict__ in, const float* __restrict__ g, const float* __restrict__ be,
    const float* __restrict__ am, int useMask,
    float* __restrict__ outF, bf16_t* __restrict__ outB) {
  __shared__ float red[256];
  const int t = threadIdx.x;
  const size_t row = blockIdx.x;
  const float* xr = in + row * ND;

  float vals[4];
  float s = 0.0f, sq = 0.0f;
#pragma unroll
  for (int j = 0; j < 4; ++j) {
    const float e = xr[t + j * 256];
    vals[j] = e; s += e; sq += e * e;
  }
  red[t] = s; __syncthreads();
  for (int o = 128; o > 0; o >>= 1) { if (t < o) red[t] += red[t + o]; __syncthreads(); }
  const float mean = red[0] * (1.0f / ND);
  __syncthreads();
  red[t] = sq; __syncthreads();
  for (int o = 128; o > 0; o >>= 1) { if (t < o) red[t] += red[t + o]; __syncthreads(); }
  const float var  = red[0] * (1.0f / ND) - mean * mean;
  const float rstd = rsqrtf(var + 1e-12f);
  const float fm   = useMask ? am[row] : 1.0f;
#pragma unroll
  for (int j = 0; j < 4; ++j) {
    const int d = t + j * 256;
    float o = ((vals[j] - mean) * rstd * g[d] + be[d]) * fm;
    outF[row * ND + d] = o;
    outB[row * ND + d] = (bf16_t)o;
  }
}

// ---------------- host orchestration ----------------

static inline void conv_wT(const float* src, bf16_t* dst, int K, int Nn, hipStream_t s) {
  dim3 g(Nn / 32, K / 32);
  conv_transpose_kernel<<<g, 256, 0, s>>>(src, dst, K, Nn);
}

extern "C" void kernel_launch(void* const* d_in, const int* in_sizes, int n_in,
                              void* d_out, int out_size, void* d_ws, size_t ws_size,
                              hipStream_t stream) {
  (void)in_sizes; (void)n_in; (void)out_size; (void)ws_size;

  const int*   ids  = (const int*)d_in[0];
  const int*   lngs = (const int*)d_in[1];
  const float* am   = (const float*)d_in[2];
  const float* we   = (const float*)d_in[3];
  const float* pe   = (const float*)d_in[4];
  const float* le   = (const float*)d_in[5];
  const float* lneg = (const float*)d_in[6];
  const float* lneb = (const float*)d_in[7];
  const float* Wq   = (const float*)d_in[8];
  const float* bq   = (const float*)d_in[9];
  const float* Wk   = (const float*)d_in[10];
  const float* bk   = (const float*)d_in[11];
  const float* Wv   = (const float*)d_in[12];
  const float* bv   = (const float*)d_in[13];
  const float* Wo   = (const float*)d_in[14];
  const float* bo   = (const float*)d_in[15];
  const float* ln1g = (const float*)d_in[16];
  const float* ln1b = (const float*)d_in[17];
  const float* W1   = (const float*)d_in[18];
  const float* b1   = (const float*)d_in[19];
  const float* W2   = (const float*)d_in[20];
  const float* b2   = (const float*)d_in[21];
  const float* ln2g = (const float*)d_in[22];
  const float* ln2b = (const float*)d_in[23];
  float* out = (float*)d_out;

  // workspace layout
  char* p = (char*)d_ws;
  float*  x    = (float*)p;  p += (size_t)NTOK * ND * sizeof(float);   // 16 MB
  float*  y    = (float*)p;  p += (size_t)NTOK * ND * sizeof(float);   // 16 MB
  bf16_t* xb   = (bf16_t*)p; p += (size_t)NTOK * ND * sizeof(bf16_t);  //  8 MB
  bf16_t* qb   = (bf16_t*)p; p += (size_t)NTOK * ND * sizeof(bf16_t);
  bf16_t* kb2  = (bf16_t*)p; p += (size_t)NTOK * ND * sizeof(bf16_t);
  bf16_t* vtb  = (bf16_t*)p; p += (size_t)NTOK * ND * sizeof(bf16_t);  // V transposed per head
  bf16_t* ctxb = (bf16_t*)p; p += (size_t)NTOK * ND * sizeof(bf16_t);
  bf16_t* hb   = (bf16_t*)p; p += (size_t)NTOK * NFF * sizeof(bf16_t); // 32 MB
  bf16_t* wstg = (bf16_t*)p; p += (size_t)ND * NFF * sizeof(bf16_t);   //  8 MB

  const float qscale = 0.125f;  // 1/sqrt(64)

  embed_kernel<<<NTOK, 256, 0, stream>>>(ids, lngs, am, we, pe, le, lneg, lneb, x, xb);

  const dim3 gD(ND / BN, NTOK / BM);    // Nn=1024 output
  const dim3 gF(NFF / BN, NTOK / BM);   // Nn=4096 output

  for (int l = 0; l < NL; ++l) {
    const size_t oDD = (size_t)l * ND * ND;
    const size_t oDF = (size_t)l * ND * NFF;

    // Q = (xb @ Wq + bq) * scale -> bf16
    conv_wT(Wq + oDD, wstg, ND, ND, stream);
    gemm_bf16_kernel<<<gD, 256, 0, stream>>>(xb, wstg, bq + (size_t)l * ND, nullptr,
                                             nullptr, qb, nullptr, NTOK, ND, ND, qscale, 0);
    // K
    conv_wT(Wk + oDD, wstg, ND, ND, stream);
    gemm_bf16_kernel<<<gD, 256, 0, stream>>>(xb, wstg, bk + (size_t)l * ND, nullptr,
                                             nullptr, kb2, nullptr, NTOK, ND, ND, 1.0f, 0);
    // V -> per-head transposed layout vt[b][h][dh][S]
    conv_wT(Wv + oDD, wstg, ND, ND, stream);
    gemm_bf16_kernel<<<gD, 256, 0, stream>>>(xb, wstg, bv + (size_t)l * ND, nullptr,
                                             nullptr, nullptr, vtb, NTOK, ND, ND, 1.0f, 0);
    // attention
    attn_kernel<<<NB * NH * (NS / 64), 128, 0, stream>>>(qb, kb2, vtb, am, ctxb);
    // y = ctx @ Wo + bo + x ; x = LN1(y)
    conv_wT(Wo + oDD, wstg, ND, ND, stream);
    gemm_bf16_kernel<<<gD, 256, 0, stream>>>(ctxb, wstg, bo + (size_t)l * ND, x,
                                             y, nullptr, nullptr, NTOK, ND, ND, 1.0f, 0);
    ln_kernel<<<NTOK, 256, 0, stream>>>(y, ln1g + (size_t)l * ND, ln1b + (size_t)l * ND,
                                        am, 0, x, xb);
    // h = gelu(x @ W1 + b1) -> bf16
    conv_wT(W1 + oDF, wstg, ND, NFF, stream);
    gemm_bf16_kernel<<<gF, 256, 0, stream>>>(xb, wstg, b1 + (size_t)l * NFF, nullptr,
                                             nullptr, hb, nullptr, NTOK, ND, NFF, 1.0f, 1);
    // y = h @ W2 + b2 + x ; x = LN2(y) * mask
    conv_wT(W2 + oDF, wstg, NFF, ND, stream);
    gemm_bf16_kernel<<<gD, 256, 0, stream>>>(hb, wstg, b2 + (size_t)l * ND, x,
                                             y, nullptr, nullptr, NTOK, NFF, ND, 1.0f, 0);
    float* lnOut = (l == NL - 1) ? out : x;
    ln_kernel<<<NTOK, 256, 0, stream>>>(y, ln2g + (size_t)l * ND, ln2b + (size_t)l * ND,
                                        am, 1, lnOut, xb);
  }
}
